// MultiHeadAttention_67473936220795
// MI455X (gfx1250) — matrix operand order repaired and checked by reading, not compiled, not run on hardware
//
#include <hip/hip_runtime.h>

// ---------------------------------------------------------------------------
// MHA for MI455X (gfx1250, wave32).  GEMMs via v_wmma_f32_16x16x32_bf16;
// pure-bf16 tile fills via TENSOR_LOAD_TO_LDS (TDM) with LDS-pad descriptors.
// d_out layout: [ out: 4096*1024 fp32 | attn: 16*4096*4096 fp32 ]
// d_ws layout : [ Qb | Kb | Vb | Cb(ctx) | Vt ]  (bf16, 8 MiB each = 40 MiB)
// Block = 128 threads (4 waves); block tile 128x64; wave strip 32x64; BK=64.
// Fragments: 2x ds_load_b128 each (row stride 72 halves = 144 B, 16B aligned).
// ---------------------------------------------------------------------------

#define SEQ     4096
#define DMODEL  1024
#define NHEADS  16
#define DHEAD   64
#define INV_TAU 0.125f

typedef __attribute__((ext_vector_type(16))) __bf16 v16bf;
typedef __attribute__((ext_vector_type(8)))  float  v8f;
typedef __attribute__((ext_vector_type(4)))  unsigned int v4u;
typedef __attribute__((ext_vector_type(8)))  int v8i_;
typedef __attribute__((ext_vector_type(4)))  int v4i_;

static constexpr int LDST = 72;      // LDS row stride in halves (144 B, 16B-aligned)

union FragBF { v16bf v; uint4 q[2]; };

__device__ __forceinline__ unsigned short f2bf(float x) {
    unsigned u = __float_as_uint(x);
    unsigned r = u + 0x7FFFu + ((u >> 16) & 1u);   // round-to-nearest-even
    return (unsigned short)(r >> 16);
}

__device__ __forceinline__ void store4bf(unsigned short* p, float4 xv) {
    uint2 d;
    d.x = ((unsigned)f2bf(xv.x)) | (((unsigned)f2bf(xv.y)) << 16);
    d.y = ((unsigned)f2bf(xv.z)) | (((unsigned)f2bf(xv.w)) << 16);
    *(uint2*)p = d;
}

// ---------------------------------------------------------------------------
// TDM: async copy of a 2D bf16 tile (tile_d0 cols x tile_d1 rows) from a
// row-major tensor (row length tensor_d0, tensor_d1 rows, line stride
// stride0 elements) into LDS at lds_addr, padding +4 DWORDs every 32 DWORDs
// so each 64-half tile row lands on a 72-half LDS stride.
// D# per CDNA5 ISA (group0/group1; groups 2,3 zero = 2D).
// This toolchain's builtin is the 6-arg form:
//   (uint32x4 g0, int32x8 g1, int32x4 g2, int32x4 g3, int32x8 pad, i32 cpol)
// ---------------------------------------------------------------------------
__device__ __forceinline__ void tdm_load_2d(unsigned lds_addr, const void* gptr,
                                            unsigned tensor_d0, unsigned tensor_d1,
                                            unsigned stride0,
                                            unsigned tile_d0, unsigned tile_d1) {
    unsigned long long ga = (unsigned long long)(size_t)gptr;   // byte address
    v4u g0;
    g0.x = 1u;                                                  // count=1 (user D#)
    g0.y = lds_addr;                                            // LDS byte address
    g0.z = (unsigned)(ga & 0xFFFFFFFFu);                        // global_addr[31:0]
    g0.w = (unsigned)((ga >> 32) & 0x01FFFFFFu) | (2u << 30);   // addr[56:32], type=2
    v8i_ g1;
    // data_size=1 (2B) | pad_enable | pad_interval=4 (32 DW) | pad_amount=3 (4 DW)
    g1[0] = (int)((1u << 16) | (1u << 20) | (4u << 22) | (3u << 25));
    g1[1] = (int)((tensor_d0 & 0xFFFFu) << 16);                 // tensor_dim0[15:0]
    g1[2] = (int)(((tensor_d0 >> 16) & 0xFFFFu) |               // tensor_dim0[31:16]
                  ((tensor_d1 & 0xFFFFu) << 16));               // tensor_dim1[15:0]
    g1[3] = (int)(((tensor_d1 >> 16) & 0xFFFFu) |               // tensor_dim1[31:16]
                  ((tile_d0 & 0xFFFFu) << 16));                 // tile_dim0
    g1[4] = (int)(tile_d1 & 0xFFFFu);                           // tile_dim1; tile_dim2=0
    g1[5] = (int)stride0;                                       // dim0_stride[31:0]
    g1[6] = 0;                                                  // stride hi / dim1_stride
    g1[7] = 0;
    v4i_ gz4 = {0, 0, 0, 0};
    v8i_ gz8 = {0, 0, 0, 0, 0, 0, 0, 0};
    __builtin_amdgcn_tensor_load_to_lds(g0, g1, gz4, gz4, gz8, 0);
}

__device__ __forceinline__ unsigned lds_off(const void* p) {
    return (unsigned)(size_t)p;        // LDS pointer -> byte offset
}

// A-matrix 16x32 bf16 fragment (ISA layout): lane L: m=L&15, K-base=(L<16)?0:8.
__device__ __forceinline__ v16bf load_frag_a(const unsigned short* lds, int mbase, int lane) {
    const unsigned short* p = lds + (mbase + (lane & 15)) * LDST + ((lane < 16) ? 0 : 8);
    FragBF f;
    f.q[0] = *(const uint4*)(p);
    f.q[1] = *(const uint4*)(p + 16);
    return f.v;
}

// B-matrix 32x16 bf16 fragment: lane L: n=L&15, K-base=(L<16)?0:8.
__device__ __forceinline__ v16bf load_frag_b(const unsigned short* lds, int nbase, int lane) {
    const unsigned short* p = lds + (nbase + (lane & 15)) * LDST + ((lane < 16) ? 0 : 8);
    FragBF f;
    f.q[0] = *(const uint4*)(p);
    f.q[1] = *(const uint4*)(p + 16);
    return f.v;
}

__device__ __forceinline__ v8f bfwmma(v16bf a, v16bf b, v8f c) {
    return __builtin_amdgcn_wmma_f32_16x16x32_bf16(false, a, false, b, (short)0, c, false, false);
}

// One BK=64 step for a wave's 32x64 strip: per substep load ALL 6 fragments
// first (12 ds_load_b128, one wait), then 8 back-to-back WMMAs.
__device__ __forceinline__ void wave_mma_k64(const unsigned short* ldsA,
                                             const unsigned short* ldsB,
                                             int wave, int lane, v8f (&acc)[2][4]) {
#pragma unroll
    for (int ks = 0; ks < 2; ++ks) {
        const unsigned short* A = ldsA + ks * 32;
        const unsigned short* B = ldsB + ks * 32;
        v16bf a0 = load_frag_a(A, wave * 32 + 0,  lane);
        v16bf a1 = load_frag_a(A, wave * 32 + 16, lane);
        v16bf b0 = load_frag_b(B, 0,  lane);
        v16bf b1 = load_frag_b(B, 16, lane);
        v16bf b2 = load_frag_b(B, 32, lane);
        v16bf b3 = load_frag_b(B, 48, lane);
        acc[0][0] = bfwmma(a0, b0, acc[0][0]);
        acc[1][0] = bfwmma(a1, b0, acc[1][0]);
        acc[0][1] = bfwmma(a0, b1, acc[0][1]);
        acc[1][1] = bfwmma(a1, b1, acc[1][1]);
        acc[0][2] = bfwmma(a0, b2, acc[0][2]);
        acc[1][2] = bfwmma(a1, b2, acc[1][2]);
        acc[0][3] = bfwmma(a0, b3, acc[0][3]);
        acc[1][3] = bfwmma(a1, b3, acc[1][3]);
    }
}

// ---------------------------------------------------------------------------
// 1) Y(bf16) = X(fp32) @ W(fp32)^T + bias ; grid (N/64, M/128), block 128
// ---------------------------------------------------------------------------
__global__ __launch_bounds__(128)
void proj_kernel(const float* __restrict__ X, const float* __restrict__ W,
                 const float* __restrict__ bias, unsigned short* __restrict__ Y) {
    __shared__ unsigned short ldsA[128 * LDST];
    __shared__ unsigned short ldsB[64 * LDST];
    const int tid = threadIdx.x, lane = tid & 31, wave = tid >> 5;
    const int mblock = blockIdx.y * 128, nblock = blockIdx.x * 64;
    v8f acc[2][4] = {};

    for (int kk = 0; kk < DMODEL; kk += 64) {
        for (int q = tid; q < 2048; q += 128) {            // A: 128x64 fp32 -> bf16
            int r = q >> 4, c = (q & 15) << 2;
            float4 xv = *(const float4*)(X + (size_t)(mblock + r) * DMODEL + kk + c);
            store4bf(ldsA + r * LDST + c, xv);
        }
        for (int q = tid; q < 1024; q += 128) {            // B: 64x64 fp32 -> bf16
            int r = q >> 4, c = (q & 15) << 2;
            float4 wv = *(const float4*)(W + (size_t)(nblock + r) * DMODEL + kk + c);
            store4bf(ldsB + r * LDST + c, wv);
        }
        __syncthreads();
        wave_mma_k64(ldsA, ldsB, wave, lane, acc);
        __syncthreads();
    }

#pragma unroll
    for (int mt = 0; mt < 2; ++mt)
#pragma unroll
    for (int nt = 0; nt < 4; ++nt) {
        int n = nblock + nt * 16 + (lane & 15);
        float bv = bias[n];
#pragma unroll
        for (int i = 0; i < 8; ++i) {
            int row = mblock + wave * 32 + mt * 16 + i + ((lane < 16) ? 0 : 8);
            Y[(size_t)row * DMODEL + n] = f2bf(acc[mt][nt][i] + bv);
        }
    }
}

// ---------------------------------------------------------------------------
// 2) logits[h,m,n] = (Qh[m,:] . Kh[n,:]) / TAU ; K = 64 (single step)
//    Tiles filled by TDM (bf16 copies).  grid (SEQ/64, SEQ/128, H), block 128
// ---------------------------------------------------------------------------
__global__ __launch_bounds__(128)
void logits_kernel(const unsigned short* __restrict__ Qb,
                   const unsigned short* __restrict__ Kb,
                   float* __restrict__ attn) {
    __shared__ unsigned short ldsA[128 * LDST];
    __shared__ unsigned short ldsB[64 * LDST];
    const int tid = threadIdx.x, lane = tid & 31, wave = tid >> 5;
    const int h = blockIdx.z;
    const int mblock = blockIdx.y * 128, nblock = blockIdx.x * 64;
    v8f acc[2][4] = {};

    if (wave == 0) {
        tdm_load_2d(lds_off(ldsA), Qb + (size_t)mblock * DMODEL + h * DHEAD,
                    DMODEL, SEQ, DMODEL, 64, 128);
        tdm_load_2d(lds_off(ldsB), Kb + (size_t)nblock * DMODEL + h * DHEAD,
                    DMODEL, SEQ, DMODEL, 64, 64);
        __builtin_amdgcn_s_wait_tensorcnt(0);
    }
    __syncthreads();
    wave_mma_k64(ldsA, ldsB, wave, lane, acc);

#pragma unroll
    for (int mt = 0; mt < 2; ++mt)
#pragma unroll
    for (int nt = 0; nt < 4; ++nt) {
        int n = nblock + nt * 16 + (lane & 15);
#pragma unroll
        for (int i = 0; i < 8; ++i) {
            int row = mblock + wave * 32 + mt * 16 + i + ((lane < 16) ? 0 : 8);
            attn[((size_t)h * SEQ + row) * SEQ + n] = acc[mt][nt][i] * INV_TAU;
        }
    }
}

// ---------------------------------------------------------------------------
// 3) In-place row softmax over attn rows of length SEQ. grid = H*SEQ blocks.
// ---------------------------------------------------------------------------
__global__ __launch_bounds__(256)
void softmax_kernel(float* __restrict__ attn) {
    __shared__ float red[256];
    float* p = attn + (size_t)blockIdx.x * SEQ;
    const int tid = threadIdx.x;

    float m = -3.4e38f;
    for (int j = tid; j < SEQ; j += 256) m = fmaxf(m, p[j]);
    red[tid] = m; __syncthreads();
    for (int s = 128; s > 0; s >>= 1) {
        if (tid < s) red[tid] = fmaxf(red[tid], red[tid + s]);
        __syncthreads();
    }
    m = red[0]; __syncthreads();

    float sum = 0.f;
    for (int j = tid; j < SEQ; j += 256) sum += __expf(p[j] - m);
    red[tid] = sum; __syncthreads();
    for (int s = 128; s > 0; s >>= 1) {
        if (tid < s) red[tid] += red[tid + s];
        __syncthreads();
    }
    float inv = 1.f / red[0];

    for (int j = tid; j < SEQ; j += 256) p[j] = __expf(p[j] - m) * inv;
}

// ---------------------------------------------------------------------------
// 3b) Vt[d][j] = Vb[j][d]  (bf16 transpose so ctx B-fill is contiguous)
// ---------------------------------------------------------------------------
__global__ __launch_bounds__(256)
void transpose_v(const unsigned short* __restrict__ Vb, unsigned short* __restrict__ Vt) {
    __shared__ unsigned short t[64][65];
    const int jt = blockIdx.x * 64, dt = blockIdx.y * 64;
    for (int idx = threadIdx.x; idx < 64 * 64; idx += 256) {
        int r = idx >> 6, c = idx & 63;
        t[r][c] = Vb[(size_t)(jt + r) * DMODEL + dt + c];
    }
    __syncthreads();
    for (int idx = threadIdx.x; idx < 64 * 64; idx += 256) {
        int r = idx >> 6, c = idx & 63;
        Vt[(size_t)(dt + r) * SEQ + jt + c] = t[c][r];
    }
}

// ---------------------------------------------------------------------------
// 4) ctx[m, h*64+n] = sum_k attn[h,m,k] * Vt[h*64+n, k] ; K = SEQ
//    B tile by TDM (overlaps the VALU fp32->bf16 A fill). grid (SEQ/128, H)
// ---------------------------------------------------------------------------
__global__ __launch_bounds__(128)
void ctx_kernel(const float* __restrict__ attn,
                const unsigned short* __restrict__ Vt,
                unsigned short* __restrict__ Cb) {
    __shared__ unsigned short ldsA[128 * LDST];
    __shared__ unsigned short ldsB[64 * LDST];
    const int tid = threadIdx.x, lane = tid & 31, wave = tid >> 5;
    const int h = blockIdx.y;
    const int mblock = blockIdx.x * 128;
    const float* Ah = attn + (size_t)h * SEQ * SEQ;
    v8f acc[2][4] = {};

    for (int kk = 0; kk < SEQ; kk += 64) {
        if (wave == 0)                                      // async B tile: 64x64 bf16
            tdm_load_2d(lds_off(ldsB), Vt + (size_t)(h * DHEAD) * SEQ + kk,
                        SEQ, DMODEL, SEQ, 64, 64);
        for (int q = tid; q < 2048; q += 128) {             // A: 128x64 fp32 -> bf16
            int r = q >> 4, c = (q & 15) << 2;
            float4 av = *(const float4*)(Ah + (size_t)(mblock + r) * SEQ + kk + c);
            store4bf(ldsA + r * LDST + c, av);
        }
        if (wave == 0) __builtin_amdgcn_s_wait_tensorcnt(0);
        __syncthreads();
        wave_mma_k64(ldsA, ldsB, wave, lane, acc);
        __syncthreads();
    }

#pragma unroll
    for (int mt = 0; mt < 2; ++mt)
#pragma unroll
    for (int nt = 0; nt < 4; ++nt) {
        int n = h * DHEAD + nt * 16 + (lane & 15);
#pragma unroll
        for (int i = 0; i < 8; ++i) {
            int row = mblock + wave * 32 + mt * 16 + i + ((lane < 16) ? 0 : 8);
            Cb[(size_t)row * DMODEL + n] = f2bf(acc[mt][nt][i]);
        }
    }
}

// ---------------------------------------------------------------------------
// 5) out(fp32) = ctx(bf16) @ Wo^T + bo ; A tile by TDM. grid (N/64, M/128)
// ---------------------------------------------------------------------------
__global__ __launch_bounds__(128)
void outproj_kernel(const unsigned short* __restrict__ Cb,
                    const float* __restrict__ W, const float* __restrict__ bias,
                    float* __restrict__ out) {
    __shared__ unsigned short ldsA[128 * LDST];
    __shared__ unsigned short ldsB[64 * LDST];
    const int tid = threadIdx.x, lane = tid & 31, wave = tid >> 5;
    const int mblock = blockIdx.y * 128, nblock = blockIdx.x * 64;
    v8f acc[2][4] = {};

    for (int kk = 0; kk < DMODEL; kk += 64) {
        if (wave == 0)                                      // async A tile: 128x64 bf16
            tdm_load_2d(lds_off(ldsA), Cb + (size_t)mblock * DMODEL + kk,
                        DMODEL, SEQ, DMODEL, 64, 128);
        for (int q = tid; q < 1024; q += 128) {             // B: 64x64 fp32 -> bf16
            int r = q >> 4, c = (q & 15) << 2;
            float4 wv = *(const float4*)(W + (size_t)(nblock + r) * DMODEL + kk + c);
            store4bf(ldsB + r * LDST + c, wv);
        }
        if (wave == 0) __builtin_amdgcn_s_wait_tensorcnt(0);
        __syncthreads();
        wave_mma_k64(ldsA, ldsB, wave, lane, acc);
        __syncthreads();
    }

#pragma unroll
    for (int mt = 0; mt < 2; ++mt)
#pragma unroll
    for (int nt = 0; nt < 4; ++nt) {
        int n = nblock + nt * 16 + (lane & 15);
        float bv = bias[n];
#pragma unroll
        for (int i = 0; i < 8; ++i) {
            int row = mblock + wave * 32 + mt * 16 + i + ((lane < 16) ? 0 : 8);
            out[(size_t)row * DMODEL + n] = acc[mt][nt][i] + bv;
        }
    }
}

// ---------------------------------------------------------------------------
extern "C" void kernel_launch(void* const* d_in, const int* in_sizes, int n_in,
                              void* d_out, int out_size, void* d_ws, size_t ws_size,
                              hipStream_t stream) {
    const float* q    = (const float*)d_in[0];
    const float* k    = (const float*)d_in[1];
    const float* v    = (const float*)d_in[2];
    const float* wq_w = (const float*)d_in[3];
    const float* wq_b = (const float*)d_in[4];
    const float* wk_w = (const float*)d_in[5];
    const float* wk_b = (const float*)d_in[6];
    const float* wv_w = (const float*)d_in[7];
    const float* wv_b = (const float*)d_in[8];
    const float* wo_w = (const float*)d_in[9];
    const float* wo_b = (const float*)d_in[10];

    float* out  = (float*)d_out;
    float* attn = out + (size_t)SEQ * DMODEL;

    const size_t MAT = (size_t)SEQ * DMODEL;
    unsigned short* Qb = (unsigned short*)d_ws;
    unsigned short* Kb = Qb + MAT;
    unsigned short* Vb = Kb + MAT;
    unsigned short* Cb = Vb + MAT;
    unsigned short* Vt = Cb + MAT;

    dim3 blk(128);
    dim3 gProj(DMODEL / 64, SEQ / 128);                    // (16, 32)
    proj_kernel<<<gProj, blk, 0, stream>>>(q, wq_w, wq_b, Qb);
    proj_kernel<<<gProj, blk, 0, stream>>>(k, wk_w, wk_b, Kb);
    proj_kernel<<<gProj, blk, 0, stream>>>(v, wv_w, wv_b, Vb);

    dim3 gLog(SEQ / 64, SEQ / 128, NHEADS);                // (64, 32, 16)
    logits_kernel<<<gLog, blk, 0, stream>>>(Qb, Kb, attn);

    softmax_kernel<<<NHEADS * SEQ, 256, 0, stream>>>(attn);

    transpose_v<<<dim3(SEQ / 64, DMODEL / 64), 256, 0, stream>>>(Vb, Vt);

    dim3 gCtx(SEQ / 128, NHEADS);                          // (32, 16)
    ctx_kernel<<<gCtx, blk, 0, stream>>>(attn, Vt, Cb);

    outproj_kernel<<<gProj, blk, 0, stream>>>(Cb, wo_w, wo_b, out);
}
